// RNNNativeModel_29454885716259
// MI455X (gfx1250) — matrix operand order repaired and checked by reading, not compiled, not run on hardware
//
#include <hip/hip_runtime.h>
#include <hip/hip_bf16.h>

typedef __attribute__((ext_vector_type(16))) __bf16 v16bf;
typedef __attribute__((ext_vector_type(8)))  float  v8f;
typedef __attribute__((ext_vector_type(4)))  int    v4i;

#define T_LEN 4096
#define HIDD  512
#define NCLS  4
#define BPAD  16                 // LDS row pad (halves): row stride 1056B = 33*32B
#define BROW  (HIDD + BPAD)

// ---- CDNA5 async global->LDS copy (builtin confirmed present; exact param
//      types from the round-2 diagnostic: arg0 = int4 AS(1)*, arg1 = LDS ptr) ----
#if defined(__gfx1250__) && __has_builtin(__builtin_amdgcn_global_load_async_to_lds_b128)
#define HAVE_ASYNC_LDS 1
typedef v4i __attribute__((address_space(1)))* gv4i_p;
typedef v4i __attribute__((address_space(3)))* lv4i_p;
#else
#define HAVE_ASYNC_LDS 0
#endif

__device__ __forceinline__ void copy16_g2lds(const void* gsrc, void* ldst) {
#if HAVE_ASYNC_LDS
    __builtin_amdgcn_global_load_async_to_lds_b128((gv4i_p)gsrc, (lv4i_p)ldst, 0, 0);
#else
    *(int4*)ldst = *(const int4*)gsrc;
#endif
}

__device__ __forceinline__ void wait_async_copies() {
#if HAVE_ASYNC_LDS
#if __has_builtin(__builtin_amdgcn_s_wait_asynccnt)
    __builtin_amdgcn_s_wait_asynccnt(0);
#else
    asm volatile("s_wait_asynccnt 0x0" ::: "memory");
#endif
#endif
}

__device__ __forceinline__ v8f wmma_bf16(v16bf a, v16bf b, v8f c) {
    return __builtin_amdgcn_wmma_f32_16x16x32_bf16(false, a, false, b,
                                                   (short)0, c, false, false);
}

// -----------------------------------------------------------------------------
// Kernel 0 (prep): one cheap pass (2 MB read / 1 MB write) that converts
//   wxT[n][k] = bf16(wx[k][n])   (K-contiguous per column -> B frags contiguous)
//   whF[...]  = wh in exact WMMA B-fragment order per (wave-tile, k-chunk, lane)
// -----------------------------------------------------------------------------
__global__ __launch_bounds__(256) void prep_kernel(
    const float* __restrict__ wx, const float* __restrict__ wh,
    __bf16* __restrict__ wxT, __bf16* __restrict__ whF)
{
    const int idx = blockIdx.x * 256 + threadIdx.x;      // 0 .. 512*512-1
    // wx transpose+convert
    const int n = idx >> 9, k = idx & 511;
    wxT[idx] = (__bf16)wx[k * HIDD + n];
    // wh -> fragment order: idx = ((w*16 + kc)*32 + lane)*16 + i
    const int i    = idx & 15;
    const int lane = (idx >> 4) & 31;
    const int blk  = idx >> 9;                           // w*16 + kc
    const int w    = blk >> 4, kc = blk & 15;
    const int kk   = kc * 32 + (lane >> 4) * 16 + i;     // B layout K index
    const int nn   = w * 16 + (lane & 15);               // B layout N index
    whF[idx] = (__bf16)wh[kk * HIDD + nn];
}

// -----------------------------------------------------------------------------
// Kernel 1: xw[t][n] = sum_k emb[seq[t]][k] * wx[k][n]
// Grid (T/16, H/64), 128 threads. The 64-col x 512-K bf16 B panel (66 KB) is
// staged into LDS with async b128 copies; fragments are contiguous LDS loads.
// -----------------------------------------------------------------------------
__global__ __launch_bounds__(128) void xw_gemm_kernel(
    const int*    __restrict__ seq,
    const float*  __restrict__ emb,
    const __bf16* __restrict__ wxT,
    float*        __restrict__ xw)
{
    __shared__ __align__(32) __bf16 bT[64 * BROW];       // 66 KB panel
    const int tid  = threadIdx.x;
    const int lane = tid & 31, wave = tid >> 5;
    const int t0   = blockIdx.x * 16;
    const int n0b  = blockIdx.y * 64;

    // ---- async-stage B panel: 4096 x 16B chunks, 32 per thread ----
    #pragma unroll
    for (int j = 0; j < 32; ++j) {
        const int c    = tid + 128 * j;                  // chunk id
        const int rowl = c >> 6;                         // local column 0..63
        const int ch   = (c & 63) * 8;                   // halves offset in row
        copy16_g2lds(wxT + (n0b + rowl) * HIDD + ch, &bT[rowl * BROW + ch]);
    }
    wait_async_copies();
    __syncthreads();

    const int row = lane & 15;                           // A row this lane feeds
    const int hi  = lane >> 4;
    const long long tok = (long long)seq[t0 + row];
    const float* arow = emb + tok * HIDD;
    const int nl  = wave * 16 + (lane & 15);             // local column
    const __bf16* bcol = &bT[nl * BROW];

    v8f c0 = {}, c1 = {};
    #pragma unroll
    for (int kc = 0; kc < 8; ++kc) {
        #pragma unroll
        for (int u = 0; u < 2; ++u) {
            const int kb = (2 * kc + u) * 32;
            v16bf a;
            #pragma unroll
            for (int i = 0; i < 16; ++i) {
                // A 16x32 bf16: lanes0-15 K={0..15}, lanes16-31 K={16..31}
                const int ka = kb + hi * 16 + i;
                a[i] = (__bf16)arow[ka];
            }
            v16bf b = *(const v16bf*)(bcol + kb + hi * 16);  // 2x ds_load_b128
            if (u == 0) c0 = wmma_bf16(a, b, c0);
            else        c1 = wmma_bf16(a, b, c1);
        }
    }
    #pragma unroll
    for (int r = 0; r < 8; ++r)
        xw[(t0 + r + 8 * hi) * HIDD + (n0b + nl)] = c0[r] + c1[r];
}

// -----------------------------------------------------------------------------
// Kernel 2: serial scan  h = tanh(xw_t + h @ wh + bias); out = h @ wfc^T.
// One workgroup, 32 waves; wave w owns columns [16w,16w+16). wh fragments live
// in VGPRs (coalesced b128 preload from whF). h is double-buffered in LDS in
// WMMA-A bf16 layout; 16 WMMAs/step as 4 independent depth-4 chains.
// -----------------------------------------------------------------------------
__global__ __launch_bounds__(1024) void rnn_scan_kernel(
    const float*  __restrict__ xw,
    const __bf16* __restrict__ whF,
    const float*  __restrict__ bias,
    const float*  __restrict__ w_fc,
    float*        __restrict__ out)
{
    __shared__ v16bf abuf[2][32];     // [buf][kc*2 + half]: h in A-fragment order
    __shared__ float hfin[HIDD];
    __shared__ float red[128];

    const int tid  = threadIdx.x;
    const int lane = tid & 31;
    const int wave = tid >> 5;
    const int hi   = lane >> 4;

    // ---- one-time coalesced preload of wh B-fragments (32B per lane per kc) ----
    v16bf b[16];
    const __bf16* wf = whF + (wave * 16 * 32 + lane) * 16;
    #pragma unroll
    for (int kc = 0; kc < 16; ++kc)
        b[kc] = *(const v16bf*)(wf + kc * 512);

    // ---- h0 = 0 ----
    ((__bf16*)&abuf[0][0])[tid] = (__bf16)0.0f;

    const int col = wave * 16 + lane;                    // producer col (lane<16)
    const float bcol = (lane < 16) ? bias[col] : 0.0f;
    int sidx = 0;
    if (lane < 16) {                                     // A-layout slot (== col)
        const int kc2 = col >> 5, r = col & 31;
        const int half2 = r >> 4;
        const int i2 = r & 15;
        sidx = kc2 * 32 + half2 * 16 + i2;
    }
    __syncthreads();

    float hnew = 0.0f;
    for (int t = 0; t < T_LEN; ++t) {
        const int cur = t & 1, nxt = cur ^ 1;
        float xwv = 0.0f;
        if (lane < 16) {
            xwv = xw[t * HIDD + col];                    // lands under WMMA chain
            if (t + 4 < T_LEN)
                __builtin_prefetch(&xw[(t + 4) * HIDD + col], 0, 0);
        }
        v8f c0 = {}, c1 = {}, c2 = {}, c3 = {};
        #pragma unroll
        for (int q = 0; q < 4; ++q) {                    // 4 chains, depth 4
            v16bf a0 = abuf[cur][(4 * q + 0) * 2 + hi];
            v16bf a1 = abuf[cur][(4 * q + 1) * 2 + hi];
            v16bf a2 = abuf[cur][(4 * q + 2) * 2 + hi];
            v16bf a3 = abuf[cur][(4 * q + 3) * 2 + hi];
            c0 = wmma_bf16(a0, b[4 * q + 0], c0);
            c1 = wmma_bf16(a1, b[4 * q + 1], c1);
            c2 = wmma_bf16(a2, b[4 * q + 2], c2);
            c3 = wmma_bf16(a3, b[4 * q + 3], c3);
        }
        if (lane < 16) {
            // A rows replicated -> D row 0 (VGPR0, lanes0-15) is the matvec
            hnew = tanhf(xwv + bcol + ((c0[0] + c1[0]) + (c2[0] + c3[0])));
            ((__bf16*)&abuf[nxt][0])[sidx] = (__bf16)hnew;
        }
        __syncthreads();                                 // 1 barrier/step
    }

    // ---- epilogue: out[c] = sum_i h[i] * w_fc[c*H + i] ----
    if (lane < 16) hfin[col] = hnew;
    __syncthreads();
    if (tid < 128) {
        const int c = tid >> 5, l2 = tid & 31;
        float s = 0.0f;
        #pragma unroll
        for (int i = 0; i < 16; ++i)
            s += hfin[l2 + 32 * i] * w_fc[c * HIDD + l2 + 32 * i];
        red[tid] = s;
    }
    __syncthreads();
    if (tid < NCLS) {
        float s = 0.0f;
        #pragma unroll
        for (int j = 0; j < 32; ++j) s += red[tid * 32 + j];
        out[tid] = s;
    }
}

// -----------------------------------------------------------------------------
extern "C" void kernel_launch(void* const* d_in, const int* in_sizes, int n_in,
                              void* d_out, int out_size, void* d_ws, size_t ws_size,
                              hipStream_t stream)
{
    (void)in_sizes; (void)n_in; (void)out_size; (void)ws_size;
    const int*   seq  = (const int*)  d_in[0];
    const float* emb  = (const float*)d_in[1];
    const float* wx   = (const float*)d_in[2];
    const float* wh   = (const float*)d_in[3];
    const float* bias = (const float*)d_in[4];
    const float* w_fc = (const float*)d_in[5];
    float* outp = (float*)d_out;

    // workspace layout: xw (8 MB fp32) | wxT bf16 (512 KB) | whF bf16 (512 KB)
    float*  xwp = (float*)d_ws;
    __bf16* wxT = (__bf16*)((char*)d_ws + (size_t)T_LEN * HIDD * sizeof(float));
    __bf16* whF = wxT + (size_t)HIDD * HIDD;

    prep_kernel<<<(HIDD * HIDD) / 256, 256, 0, stream>>>(wx, wh, wxT, whF);
    xw_gemm_kernel<<<dim3(T_LEN / 16, HIDD / 64), 128, 0, stream>>>(seq, emb, wxT, xwp);
    rnn_scan_kernel<<<1, 1024, 0, stream>>>(xwp, whF, bias, w_fc, outp);
}
